// RegionProposalLayer_51986284151190
// MI455X (gfx1250) — matrix operand order repaired and checked by reading, not compiled, not run on hardware
//
#include <hip/hip_runtime.h>

typedef __attribute__((ext_vector_type(16))) _Float16 v16h;
typedef __attribute__((ext_vector_type(8)))  float    v8f;
typedef unsigned int u32;
typedef unsigned long long u64;

#define GP 60
#define GQ 40
#define NA 9
#define NN (GP*GQ*NA)   /* 21600 */
#define KK 1000
#define KPAD 1024
#define MAXB 18
#define NMS_T 0.5f
#define NTHREADS 256

__device__ __forceinline__ u32 sortkey(float f) {
  u32 u = __float_as_uint(f);
  return (u & 0x80000000u) ? ~u : (u | 0x80000000u);
}

__global__ void __launch_bounds__(NTHREADS)
rpn_nms_kernel(const float* __restrict__ x,
               const float* __restrict__ anchors,
               float* __restrict__ out)
{
  __shared__ u32 s_hist[256];
  __shared__ u64 s_items[KPAD];
  __shared__ float s_x1[KPAD], s_y1[KPAD], s_x2[KPAD], s_y2[KPAD], s_area[KPAD];
  __shared__ float s_cand[KPAD * 5];
  __shared__ u32 s_keep[32];          // 1024-bit keep mask
  __shared__ float s_pref[KPAD];      // per-row (16-wide) inclusive prefix of keep
  __shared__ float s_rowoff[64];
  __shared__ int s_sel[MAXB];
  __shared__ u64 s_prefix;
  __shared__ int s_rank;
  __shared__ u32 s_count;
  __shared__ int s_nkeep;

  const int tid = threadIdx.x;
  const int b = blockIdx.x;
  const float* xb = x + (size_t)b * NN * 6;

  // --- Prefetch this sample's slab (emits global_prefetch_b8 on gfx1250) ---
  {
    const char* p = (const char*)xb;
    const size_t bytes = (size_t)NN * 6 * 4;
    for (size_t off = (size_t)tid * 256; off < bytes; off += (size_t)NTHREADS * 256)
      __builtin_prefetch(p + off, 0, 1);
  }

  if (tid == 0) { s_prefix = 0ull; s_rank = KK - 1; s_count = 0u; }

  // --- Phase A: exact rank-999 threshold via 8-digit MSB radix select -------
  // 64-bit item = (sortkey(score) << 32) | ~index  (distinct; tie-break = lower idx first)
  for (int d = 7; d >= 0; --d) {
    for (int t = tid; t < 256; t += NTHREADS) s_hist[t] = 0u;
    __syncthreads();
    const u64 pre = s_prefix;
    const u64 hm = (d == 7) ? 0ull : (~0ull << (8 * (d + 1)));
    for (int i = tid; i < NN; i += NTHREADS) {
      u32 key = sortkey(xb[(size_t)i * 6]);
      u64 it = ((u64)key << 32) | (u32)(~(u32)i);
      if (((it ^ pre) & hm) == 0ull)
        atomicAdd(&s_hist[(u32)(it >> (8 * d)) & 0xFFu], 1u);
    }
    __syncthreads();
    if (tid == 0) {
      int r = s_rank;
      u32 acc = 0u;
      int bin;
      for (bin = 255; bin > 0; --bin) {
        u32 c = s_hist[bin];
        if (acc + c > (u32)r) break;
        acc += c;
      }
      s_rank = r - (int)acc;
      s_prefix = s_prefix | ((u64)(u32)bin << (8 * d));
    }
    __syncthreads();
  }

  // --- Phase B: compact the exactly-1000 items >= threshold ------------------
  const u64 T = s_prefix;
  for (int i = tid; i < NN; i += NTHREADS) {
    u32 key = sortkey(xb[(size_t)i * 6]);
    u64 it = ((u64)key << 32) | (u32)(~(u32)i);
    if (it >= T) {
      u32 pos = atomicAdd(&s_count, 1u);
      if (pos < (u32)KPAD) s_items[pos] = it;
    }
  }
  for (int t = KK + tid; t < KPAD; t += NTHREADS) s_items[t] = 0ull;

  // --- Phase C: bitonic sort 1024 items descending (order == top_k order) ----
  for (u32 ksz = 2; ksz <= (u32)KPAD; ksz <<= 1) {
    for (u32 j = ksz >> 1; j > 0; j >>= 1) {
      __syncthreads();
      for (u32 q = tid; q < (u32)(KPAD / 2); q += NTHREADS) {
        u32 i1 = 2u * q - (q & (j - 1u));
        u32 i2 = i1 | j;
        u64 va = s_items[i1], vb = s_items[i2];
        bool desc = ((i1 & ksz) == 0u);
        if (desc ? (va < vb) : (va > vb)) { s_items[i1] = vb; s_items[i2] = va; }
      }
    }
  }
  __syncthreads();

  // --- Phase D: decode boxes with anchors; init keep mask --------------------
  if (tid < 32) s_keep[tid] = (tid < 31) ? 0xFFFFFFFFu : 0x000000FFu;  // 1000 bits
  for (int t = tid; t < KK; t += NTHREADS) {
    u64 it = s_items[t];
    u32 idx = ~(u32)(it & 0xFFFFFFFFull);
    u32 key = (u32)(it >> 32);
    u32 fb = (key & 0x80000000u) ? (key & 0x7FFFFFFFu) : ~key;
    float score = __uint_as_float(fb);
    int p = ((int)idx / NA) % GP;
    int q = (int)idx / (GP * NA);
    int sr = (int)idx % NA;
    const float* a = anchors + (((size_t)(p * GQ + q) * NA + sr) << 2);
    const float* xr = xb + (size_t)idx * 6;
    float r0 = xr[2], r1 = xr[3], r2 = xr[4], r3 = xr[5];
    float a0 = a[0], a1 = a[1], a2 = a[2], a3 = a[3];
    float xc = r0 * a2 + a0;
    float yc = r1 * a3 + a1;
    float w = a2 * expf(r2);
    float h = a3 * expf(r3);
    float x1v = xc - 0.5f * w, x2v = xc + 0.5f * w;
    float y1v = yc - 0.5f * h, y2v = yc + 0.5f * h;
    s_cand[t * 5 + 0] = xc;
    s_cand[t * 5 + 1] = yc;
    s_cand[t * 5 + 2] = w;
    s_cand[t * 5 + 3] = h;
    s_cand[t * 5 + 4] = score;
    s_x1[t] = x1v; s_y1[t] = y1v; s_x2[t] = x2v; s_y2[t] = y2v;
    s_area[t] = (x2v - x1v) * (y2v - y1v);
  }
  __syncthreads();

  // Cache this thread's j-boxes in registers (j = tid + 256k)
  float rx1[4], ry1[4], rx2[4], ry2[4], rar[4];
#pragma unroll
  for (int k = 0; k < 4; ++k) {
    int jj = tid + k * NTHREADS;
    if (jj < KK) {
      rx1[k] = s_x1[jj]; ry1[k] = s_y1[jj];
      rx2[k] = s_x2[jj]; ry2[k] = s_y2[jj];
      rar[k] = s_area[jj];
    } else {
      rx1[k] = 0.f; ry1[k] = 0.f; rx2[k] = 0.f; ry2[k] = 0.f; rar[k] = 1.f;
    }
  }

  // --- Phase E: greedy NMS (suppress j>i where inter/area[j] > 0.5) ----------
  for (int i = 0; i < KK; ++i) {
    bool kept = ((s_keep[i >> 5] >> (i & 31)) & 1u) != 0u;  // uniform broadcast
    if (!kept) continue;
    float ix1 = s_x1[i], iy1 = s_y1[i], ix2 = s_x2[i], iy2 = s_y2[i];
#pragma unroll
    for (int k = 0; k < 4; ++k) {
      int jj = tid + k * NTHREADS;
      if (jj > i && jj < KK) {
        float xx1 = fmaxf(ix1, rx1[k]);
        float yy1 = fmaxf(iy1, ry1[k]);
        float xx2 = fminf(ix2, rx2[k]);
        float yy2 = fminf(iy2, ry2[k]);
        float iw = fmaxf(0.0f, xx2 - xx1);
        float ih = fmaxf(0.0f, yy2 - yy1);
        if (iw * ih / rar[k] > NMS_T)
          atomicAnd(&s_keep[jj >> 5], ~(1u << (jj & 31)));
      }
    }
    __syncthreads();
  }
  __syncthreads();

  // --- Phase F: ranks = cumsum(keep)-1 via WMMA scan --------------------------
  // View keep[0..1023] as 64x16 matrix R (row r = bits 16r..16r+15).
  // Per 16-row tile: D = A * U, A = keep tile (K=0..15, zero-padded to K=32),
  // U upper-triangular ones => D[m][n] = inclusive prefix within row.
  {
    const int lane = tid & 31;
    const int wv = tid >> 5;
    if (wv < 4) {
      const int Mrow = lane & 15;
      const int hi = (lane >= 16) ? 1 : 0;
      v16h a, bm;
#pragma unroll
      for (int h = 0; h < 16; ++h) {
        int v = h >> 1;
        int Ka = ((v < 4) ? (2 * v) : (16 + 2 * (v - 4))) + (h & 1) + hi * 8;
        float av = 0.0f;
        if (Ka < 16) {
          int e = (16 * wv + Mrow) * 16 + Ka;
          av = (float)((s_keep[e >> 5] >> (e & 31)) & 1u);
        }
        a[h] = (_Float16)av;
        int Kb = h + hi * 8 + ((h >= 8) ? 8 : 0);  // B: N = lane%16, K = table order (same as A)
        bm[h] = (_Float16)((Kb < 16 && Kb <= Mrow) ? 1.0f : 0.0f);
      }
      v8f c = {};
      v8f dd = __builtin_amdgcn_wmma_f32_16x16x32_f16(
          false, a, false, bm, (short)0, c, false, false);
#pragma unroll
      for (int g = 0; g < 8; ++g) {
        int Mr = g + hi * 8;                // D: row = g (+8 for hi lanes), col = lane%16
        s_pref[(16 * wv + Mr) * 16 + (lane & 15)] = dd[g];
      }
    }
  }
  __syncthreads();

  if (tid == 0) {
    float off = 0.0f;
    for (int r = 0; r < 64; ++r) { s_rowoff[r] = off; off += s_pref[r * 16 + 15]; }
    s_nkeep = (int)off;
  }
  __syncthreads();

  for (int t = tid; t < KK; t += NTHREADS) {
    if ((s_keep[t >> 5] >> (t & 31)) & 1u) {
      int rank = (int)(s_rowoff[t >> 4] + s_pref[t]) - 1;
      if (rank < MAXB) s_sel[rank] = t;
    }
  }
  __syncthreads();

  // --- Phase G: emit 18x5 with the reference's padding rule -------------------
  if (tid < MAXB * 5) {
    int j = tid / 5, c = tid % 5;
    int n = (s_nkeep < MAXB) ? s_nkeep : MAXB;
    int src;
    if (j < n) {
      src = s_sel[j];
    } else {
      int pidx = j - n;                     // clip(j-n, 0, K-1); j>=n so >=0
      src = (pidx < (KK - 1)) ? pidx : (KK - 1);
    }
    out[(size_t)b * (MAXB * 5) + tid] = s_cand[src * 5 + c];
  }
}

extern "C" void kernel_launch(void* const* d_in, const int* in_sizes, int n_in,
                              void* d_out, int out_size, void* d_ws, size_t ws_size,
                              hipStream_t stream) {
  (void)n_in; (void)out_size; (void)d_ws; (void)ws_size;
  const float* x = (const float*)d_in[0];
  const float* anchors = (const float*)d_in[1];
  float* out = (float*)d_out;
  int batch = in_sizes[0] / (NN * 6);       // 64
  rpn_nms_kernel<<<batch, NTHREADS, 0, stream>>>(x, anchors, out);
}
